// TitansMemory_74457553044432
// MI455X (gfx1250) — matrix operand order repaired and checked by reading, not compiled, not run on hardware
//
#include <hip/hip_runtime.h>
#include <math.h>

// Problem constants: B=4, T=8192 -> N=32768 rows, D=1024, M=512
#define D_DIM 1024
#define M_DIM 512
#define ROWS_WG 64           // rows of h per workgroup (4 waves x 16 rows)
#define BLOCK 128
#define HS 1032              // h-tile row stride (bf16 elems): 2064B rows, conflict-free
#define SP 520               // P row stride (bf16 elems): 1040B rows, conflict-free
#define EPSN 1e-12f

typedef __bf16 bf16;
typedef __bf16 v4bf  __attribute__((ext_vector_type(4)));
typedef __bf16 v8bf  __attribute__((ext_vector_type(8)));
typedef __bf16 v16bf __attribute__((ext_vector_type(16)));
typedef float  v4f   __attribute__((ext_vector_type(4)));
typedef float  v8f   __attribute__((ext_vector_type(8)));

// LDS byte offsets
#define OFF_HT 0
#define SZ_HT  (ROWS_WG * HS * 2)          // 132096
#define OFF_P  (OFF_HT + SZ_HT)
#define SZ_P   (ROWS_WG * SP * 2)          // 66560
#define OFF_RN (OFF_P + SZ_P)              // 198656
#define OFF_MV (OFF_RN + ROWS_WG * 4)      // 198912
#define LDS_BYTES (OFF_MV + M_DIM * 4)     // 200960 (< 320KB)

extern __shared__ char ldsraw[];

__device__ __forceinline__ v16bf cat16(v8bf lo, v8bf hi) {
  return __builtin_shufflevector(lo, hi, 0,1,2,3,4,5,6,7,8,9,10,11,12,13,14,15);
}

// ---------------------------------------------------------------------------
// Pre-kernel 1: memb = bf16(mem) row-major; minv[m] = strength[m]/max(||mem_m||,eps)
// One wave per mem row, 8 waves per block.
// ---------------------------------------------------------------------------
__global__ void titans_memprep_kernel(const float* __restrict__ mem,
                                      const float* __restrict__ strength,
                                      bf16* __restrict__ memb,
                                      float* __restrict__ minv) {
  const int wave = threadIdx.x >> 5;
  const int lane = threadIdx.x & 31;
  const int row  = blockIdx.x * 8 + wave;
  const float* mr = mem + (size_t)row * D_DIM;
  bf16* br = memb + (size_t)row * D_DIM;
  float s = 0.f;
  #pragma unroll 8
  for (int i = lane; i < D_DIM; i += 32) {
    float v = mr[i];
    s += v * v;
    br[i] = (bf16)v;
  }
  #pragma unroll
  for (int m = 16; m >= 1; m >>= 1) s += __shfl_xor(s, m, 32);
  if (lane == 0) minv[row] = strength[row] / fmaxf(sqrtf(s), EPSN);
}

// ---------------------------------------------------------------------------
// Pre-kernel 2: memT[d][m] = bf16(mem[m][d])  (bf16 transposed copy for GEMM2 B)
// ---------------------------------------------------------------------------
__global__ void titans_memtrans_kernel(const float* __restrict__ mem,
                                       bf16* __restrict__ memT) {
  const int idx = blockIdx.x * 256 + threadIdx.x;   // over D*M
  const int d = idx >> 9;                            // / M_DIM
  const int m = idx & (M_DIM - 1);
  memT[idx] = (bf16)mem[(size_t)m * D_DIM + d];
}

// ---------------------------------------------------------------------------
// Main fused kernel (bf16 WMMA, f32 accumulate):
//   logits = (h.memb^T) * rn * minv  -> bf16 P (LDS)
//   softmax rows (f32)               -> bf16 P
//   out = P . mem  via memT
// ---------------------------------------------------------------------------
__global__ void titans_read_kernel(const float* __restrict__ h,
                                   const bf16* __restrict__ memb,
                                   const bf16* __restrict__ memT,
                                   const float* __restrict__ minv_g,
                                   float* __restrict__ out) {
  bf16*  ht    = (bf16*)(ldsraw + OFF_HT);    // [ROWS_WG][HS]
  bf16*  P     = (bf16*)(ldsraw + OFF_P);     // [ROWS_WG][SP]
  float* rnacc = (float*)(ldsraw + OFF_RN);   // [ROWS_WG]
  float* minvs = (float*)(ldsraw + OFF_MV);   // [M_DIM]

  const int tid   = threadIdx.x;
  const int wave  = tid >> 5;
  const int lane  = tid & 31;
  const int lhalf = lane & 15;
  const int hi    = lane >> 4;        // lane half: K-split selector
  const int row0  = wave * 16;
  const long wg0  = (long)blockIdx.x * ROWS_WG;

  if (tid < ROWS_WG) rnacc[tid] = 0.f;
  #pragma unroll
  for (int i = 0; i < M_DIM / BLOCK; ++i) minvs[tid + i * BLOCK] = minv_g[tid + i * BLOCK];
  __syncthreads();

  // ---- stage h (f32 -> bf16 LDS) once, accumulating ||h_row||^2 in f32 ----
  {
    const int r = tid >> 1;                 // 2 threads per row
    const int halfc = (tid & 1) * (D_DIM / 2);
    const float* hr = h + (wg0 + r) * D_DIM + halfc;
    bf16* dst = ht + r * HS + halfc;
    float s = 0.f;
    for (int i = 0; i < D_DIM / 8; ++i) {   // 128 x 4 floats = 512 cols
      v4f v = __builtin_nontemporal_load((const v4f*)(hr + i * 4));
      s += v[0]*v[0] + v[1]*v[1] + v[2]*v[2] + v[3]*v[3];
      v4bf b; b[0] = (bf16)v[0]; b[1] = (bf16)v[1]; b[2] = (bf16)v[2]; b[3] = (bf16)v[3];
      *(v4bf*)(dst + i * 4) = b;
    }
    atomicAdd(&rnacc[r], s);                // ds_add_f32
  }
  __syncthreads();
  // Everything below is intra-wave on owned rows -> no further barriers.

  // per-row normalization factors 1/max(||h_row||, eps) in D-matrix layout
  float rnv[8];
  #pragma unroll
  for (int v = 0; v < 8; ++v)
    rnv[v] = 1.f / fmaxf(sqrtf(rnacc[row0 + v + hi * 8]), EPSN);

  // ---- GEMM1: 16x16 logit tiles, full-K register accumulation --------------
  // A (16x32 bf16): lane m, elems 0..7 = K(hi*8..+7), elems 8..15 = K(16+hi*8..+7)
  // B (32x16 bf16): lane n, elems 0..7 = K(hi*8..+7), elems 8..15 = K(16+hi*8..+7)
  const bf16* arow = ht + (row0 + lhalf) * HS + hi * 8;
  for (int mt = 0; mt < M_DIM / 16; ++mt) {
    v8f acc = {};
    const bf16* brow = memb + (size_t)(mt * 16 + lhalf) * D_DIM + hi * 8;
    #pragma unroll 4
    for (int k0 = 0; k0 < D_DIM; k0 += 32) {
      v16bf a = cat16(*(const v8bf*)(arow + k0), *(const v8bf*)(arow + k0 + 16));
      v16bf b = cat16(*(const v8bf*)(brow + k0), *(const v8bf*)(brow + k0 + 16));
      acc = __builtin_amdgcn_wmma_f32_16x16x32_bf16(
          false, a, false, b, (short)0, acc, false, false);
    }
    const float mv = minvs[mt * 16 + lhalf];
    #pragma unroll
    for (int v = 0; v < 8; ++v)
      P[(row0 + v + hi * 8) * SP + mt * 16 + lhalf] = (bf16)(acc[v] * rnv[v] * mv);
  }

  // ---- softmax over M per row (f32 math, bf16 storage) ---------------------
  for (int r = 0; r < 16; ++r) {
    const int lr = row0 + r;
    float vals[16];
    float mx = -INFINITY;
    #pragma unroll
    for (int i = 0; i < 16; ++i) {
      float v = (float)P[lr * SP + lane + i * 32];
      vals[i] = v;
      mx = fmaxf(mx, v);
    }
    #pragma unroll
    for (int m = 16; m >= 1; m >>= 1) mx = fmaxf(mx, __shfl_xor(mx, m, 32));
    float sum = 0.f;
    #pragma unroll
    for (int i = 0; i < 16; ++i) {
      float e = __expf(vals[i] - mx);
      vals[i] = e;
      sum += e;
    }
    #pragma unroll
    for (int m = 16; m >= 1; m >>= 1) sum += __shfl_xor(sum, m, 32);
    const float inv = 1.f / sum;
    #pragma unroll
    for (int i = 0; i < 16; ++i)
      P[lr * SP + lane + i * 32] = (bf16)(vals[i] * inv);
  }

  // ---- GEMM2: out = P . mem (B fragments from bf16 transposed memT) --------
  const bf16* prow = P + (row0 + lhalf) * SP + hi * 8;
  for (int nt = 0; nt < D_DIM / 16; ++nt) {
    v8f acc = {};
    const bf16* brow = memT + (size_t)(nt * 16 + lhalf) * M_DIM + hi * 8;
    #pragma unroll 4
    for (int k0 = 0; k0 < M_DIM; k0 += 32) {
      v16bf a = cat16(*(const v8bf*)(prow + k0), *(const v8bf*)(prow + k0 + 16));
      v16bf b = cat16(*(const v8bf*)(brow + k0), *(const v8bf*)(brow + k0 + 16));
      acc = __builtin_amdgcn_wmma_f32_16x16x32_bf16(
          false, a, false, b, (short)0, acc, false, false);
    }
    #pragma unroll
    for (int v = 0; v < 8; ++v)
      __builtin_nontemporal_store(acc[v],
          out + (wg0 + row0 + v + hi * 8) * D_DIM + nt * 16 + lhalf);
  }
}

// ---------------------------------------------------------------------------
extern "C" void kernel_launch(void* const* d_in, const int* in_sizes, int n_in,
                              void* d_out, int out_size, void* d_ws, size_t ws_size,
                              hipStream_t stream) {
  const float* h        = (const float*)d_in[0];   // [B,T,D] fp32
  const float* mem      = (const float*)d_in[1];   // [M,D]   fp32
  const float* strength = (const float*)d_in[2];   // [M]     fp32
  float* out = (float*)d_out;                      // [B,T,D] fp32

  // workspace: memb (bf16 [M][D]) | memT (bf16 [D][M]) | minv (f32 [M])
  bf16*  memb = (bf16*)d_ws;
  bf16*  memT = memb + (size_t)M_DIM * D_DIM;
  float* minv = (float*)(memT + (size_t)D_DIM * M_DIM);

  const int nrows = in_sizes[0] / D_DIM;           // 32768

  titans_memprep_kernel<<<M_DIM / 8, 256, 0, stream>>>(mem, strength, memb, minv);
  titans_memtrans_kernel<<<(D_DIM * M_DIM) / 256, 256, 0, stream>>>(mem, memT);
  titans_read_kernel<<<nrows / ROWS_WG, BLOCK, LDS_BYTES, stream>>>(h, memb, memT, minv, out);
}
